// PointNet2_4964982194784
// MI455X (gfx1250) — matrix operand order repaired and checked by reading, not compiled, not run on hardware
//
#include <hip/hip_runtime.h>
#include <hip/hip_bf16.h>
#include <math.h>

// ---------------- constants ----------------
#define BATCH 8
#define NPTS  8192
#define NIFC  7
#define NCLS  5
#define NS1   2048
#define NS2   512
#define NS3   128
#define KNN_K 32
#define BN_INV 0.9999950000374997f   // 1/sqrt(1+1e-5)

typedef __attribute__((ext_vector_type(16))) _Float16 v16h;
typedef __attribute__((ext_vector_type(8)))  _Float16 v8h;
typedef __attribute__((ext_vector_type(8)))  float    v8f;

static inline int cdiv(int a, int b) { return (a + b - 1) / b; }
static inline int pad32(int c) { return (c + 31) & ~31; }

// ---------------- kernels ----------------

// Extract pos (f32, [B][N][3]) and feat (f16, [B][N][7]) from cloud_data [B][9][N]
__global__ void k_prepare(const float* __restrict__ cloud,
                          float* __restrict__ pos, _Float16* __restrict__ feat)
{
    int gid = blockIdx.x * blockDim.x + threadIdx.x;
    if (gid >= BATCH * NPTS) return;
    int n = gid % NPTS, b = gid / NPTS;
    const float* cd = cloud + (size_t)b * 9 * NPTS;
    float* P = pos + (size_t)gid * 3;
    P[0] = cd[0 * NPTS + n];
    P[1] = cd[1 * NPTS + n];
    P[2] = cd[2 * NPTS + n];
    _Float16* F = feat + (size_t)gid * NIFC;
#pragma unroll
    for (int c = 0; c < NIFC; ++c) F[c] = (_Float16)cd[(2 + c) * NPTS + n];
}

// Farthest point sampling: one block per cloud, distances resident in LDS.
__global__ void __launch_bounds__(1024)
k_fps(const float* __restrict__ srcpos, int ns, int nsample, int* __restrict__ outidx)
{
    __shared__ float sd[8192];
    __shared__ float rv[1024];
    __shared__ int   ri[1024];
    __shared__ int   snxt;
    int b = blockIdx.x;
    const float* P = srcpos + (size_t)b * ns * 3;
    int* idx = outidx + (size_t)b * nsample;
    int t = threadIdx.x;
    float p0x = P[0], p0y = P[1], p0z = P[2];
    for (int i = t; i < ns; i += 1024) {
        float dx = P[3*i] - p0x, dy = P[3*i+1] - p0y, dz = P[3*i+2] - p0z;
        sd[i] = dx*dx + dy*dy + dz*dz;
    }
    if (t == 0) idx[0] = 0;
    __syncthreads();
    for (int s = 1; s < nsample; ++s) {
        float bv = -1.0f; int bidx = 0x7fffffff;
        for (int i = t; i < ns; i += 1024) {
            float v = sd[i];
            if (v > bv) { bv = v; bidx = i; }
        }
        rv[t] = bv; ri[t] = bidx;
        __syncthreads();
        for (int off = 512; off > 0; off >>= 1) {
            if (t < off) {
                float v2 = rv[t + off]; int i2 = ri[t + off];
                if (v2 > rv[t] || (v2 == rv[t] && i2 < ri[t])) { rv[t] = v2; ri[t] = i2; }
            }
            __syncthreads();
        }
        if (t == 0) { snxt = ri[0]; idx[s] = ri[0]; }
        __syncthreads();
        int nx = snxt;
        float nxx = P[3*nx], nxy = P[3*nx+1], nxz = P[3*nx+2];
        for (int i = t; i < ns; i += 1024) {
            float dx = P[3*i] - nxx, dy = P[3*i+1] - nxy, dz = P[3*i+2] - nxz;
            float d = dx*dx + dy*dy + dz*dz;
            if (d < sd[i]) sd[i] = d;
        }
        __syncthreads();
    }
}

// q[b][i][:] = src[b][idx[b][i]][:]
__global__ void k_gather(const float* __restrict__ src, const int* __restrict__ idx,
                         int ns, int nsample, float* __restrict__ q)
{
    int gid = blockIdx.x * blockDim.x + threadIdx.x;
    if (gid >= BATCH * nsample) return;
    int i = gid % nsample, b = gid / nsample;
    int s = idx[(size_t)b * nsample + i];
    const float* S = src + ((size_t)b * ns + s) * 3;
    float* Q = q + (size_t)gid * 3;
    Q[0] = S[0]; Q[1] = S[1]; Q[2] = S[2];
}

// K-nearest-neighbors, one thread per query.
__global__ void k_knn(const float* __restrict__ qpos, const float* __restrict__ spos,
                      int nq, int ns, int* __restrict__ nbr, float* __restrict__ nd2)
{
    int gid = blockIdx.x * blockDim.x + threadIdx.x;
    if (gid >= BATCH * nq) return;
    int b = gid / nq;
    const float* Q = qpos + (size_t)gid * 3;
    const float* P = spos + (size_t)b * ns * 3;
    float qx = Q[0], qy = Q[1], qz = Q[2];
    float bd[KNN_K]; int bi[KNN_K];
#pragma unroll
    for (int j = 0; j < KNN_K; ++j) { bd[j] = 1e30f; bi[j] = 0; }
    float worst = 1e30f; int ws = 0;
    for (int s = 0; s < ns; ++s) {
        float dx = P[3*s] - qx, dy = P[3*s+1] - qy, dz = P[3*s+2] - qz;
        float d = dx*dx + dy*dy + dz*dz;
        if (d < worst) {
            bd[ws] = d; bi[ws] = s;
            worst = bd[0]; ws = 0;
#pragma unroll
            for (int j = 1; j < KNN_K; ++j) if (bd[j] > worst) { worst = bd[j]; ws = j; }
        }
    }
    int*   on = nbr + (size_t)gid * KNN_K;
    float* od = nd2 + (size_t)gid * KNN_K;
#pragma unroll
    for (int j = 0; j < KNN_K; ++j) { on[j] = bi[j]; od[j] = bd[j]; }
}

// msg[(b,q,j)][:] = [ xsrc[nbr], spos[nbr]-q ] as f16, row stride cinp (zero padded)
__global__ void k_build_msg(const _Float16* __restrict__ xsrc, int cs,
                            const float* __restrict__ spos, const float* __restrict__ qpos,
                            const int* __restrict__ nbr, int nq, int ns,
                            _Float16* __restrict__ A, int cinp)
{
    int gid = blockIdx.x * blockDim.x + threadIdx.x;
    int total = BATCH * nq * KNN_K;
    if (gid >= total) return;
    int qi = (gid / KNN_K) % nq;
    int b = gid / (KNN_K * nq);
    int nb = nbr[(size_t)gid];
    const _Float16* X = xsrc + ((size_t)b * ns + nb) * cs;
    _Float16* R = A + (size_t)gid * cinp;
    for (int c = 0; c < cs; ++c) R[c] = X[c];
    const float* SP = spos + ((size_t)b * ns + nb) * 3;
    const float* QP = qpos + ((size_t)b * nq + qi) * 3;
#pragma unroll
    for (int d = 0; d < 3; ++d) R[cs + d] = (_Float16)(SP[d] - QP[d]);
    for (int c = cs + 3; c < cinp; ++c) R[c] = (_Float16)0.f;
}

// Pack weights f32[cin][cout] -> WMMA B-fragment order f16, zero padded:
// Wpk[((kt*coutp + c)*2 + hs)*16 + i] = W[kt*32 + (i < 8 ? 8*hs + i : 16 + 8*hs + i - 8)][c]
__global__ void k_cvt_pack(const float* __restrict__ W, _Float16* __restrict__ Wpk,
                           int cin, int cout, int coutp, int n)
{
    int gid = blockIdx.x * blockDim.x + threadIdx.x;
    if (gid >= n) return;
    int i  = gid & 15;
    int hs = (gid >> 4) & 1;
    int c  = (gid >> 5) % coutp;
    int kt = gid / (32 * coutp);
    int k  = kt * 32 + (i < 8 ? 8 * hs + i : 16 + 8 * hs + (i - 8));
    _Float16 v = (_Float16)0.f;
    if (k < cin && c < cout) v = (_Float16)W[(size_t)k * cout + c];
    Wpk[gid] = v;
}

// WMMA GEMM + bias + (relu) + (batchnorm).  One wave computes a 16x16 D tile.
// A: f16 [M][cinp] (pad zeroed), Wpk: packed B fragments, out stride coutp (pad zeroed).
// mode: 0 = +bias only, 1 = +bias,relu, 2 = +bias,relu,bn
__global__ void __launch_bounds__(256)
k_gemm(const _Float16* __restrict__ A, const _Float16* __restrict__ Wpk,
       const float* __restrict__ bias, const float* __restrict__ gw,
       const float* __restrict__ bw, _Float16* __restrict__ outh,
       float* __restrict__ outf, int M, int cinp, int cout, int coutp, int mode)
{
    int wave = blockIdx.x * (blockDim.x >> 5) + (threadIdx.x >> 5);
    int lane = threadIdx.x & 31;
    int tilesN = coutp >> 4;
    int tilesM = M >> 4;
    if (wave >= tilesM * tilesN) return;     // wave-uniform exit
    int mt = wave / tilesN, nt = wave % tilesN;
    int hs = lane >> 4;
    int rl = lane & 15;
    int row = mt * 16 + rl;                  // A row for this lane
    int col = nt * 16 + rl;                  // B column for this lane

    const _Float16* Arow = A + (size_t)row * cinp + 8 * hs;
    const _Float16* Bcol = Wpk + ((size_t)col * 2 + hs) * 16;
    size_t bstep = (size_t)coutp * 32;

    v8f acc = {};
    int ktiles = cinp >> 5;
    for (int kt = 0; kt < ktiles; ++kt) {
        v8h alo = *(const v8h*)(Arow + kt * 32);        // K = 8*hs .. +7
        v8h ahi = *(const v8h*)(Arow + kt * 32 + 16);   // K = 16+8*hs .. +7
        v16h a = __builtin_shufflevector(alo, ahi,
                    0,1,2,3,4,5,6,7,8,9,10,11,12,13,14,15);
        v16h bvec = *(const v16h*)(Bcol + (size_t)kt * bstep);
        acc = __builtin_amdgcn_wmma_f32_16x16x32_f16(
                  false, a, false, bvec, (short)0, acc, false, false);
    }
    // epilogue: D element (row = e + 8*hs, col = rl) in acc[e]
    int cg = nt * 16 + rl;
    float bb = 0.f, gg = 1.f, ee = 0.f;
    if (cg < cout) {
        bb = bias[cg];
        if (mode == 2) { gg = gw[cg]; ee = bw[cg]; }
    }
#pragma unroll
    for (int e = 0; e < 8; ++e) {
        float v = acc[e] + bb;
        if (mode >= 1) v = fmaxf(v, 0.f);
        if (mode == 2) v = v * gg * BN_INV + ee;
        int rg = mt * 16 + 8 * hs + e;
        if (outh) outh[(size_t)rg * coutp + cg] = (cg < cout) ? (_Float16)v : (_Float16)0.f;
        if (outf && cg < cout) outf[(size_t)rg * cout + cg] = v;
    }
}

// masked max over K neighbors (radius mask), h: [B*nq*K][ldh] -> x: [B*nq][cout]
__global__ void k_sa_pool(const _Float16* __restrict__ h, const float* __restrict__ d2,
                          float r2, int nq, int cout, int ldh, _Float16* __restrict__ xout)
{
    int gid = blockIdx.x * blockDim.x + threadIdx.x;
    if (gid >= BATCH * nq * cout) return;
    int c = gid % cout;
    int bq = gid / cout;
    float m = -INFINITY;
    const float* D = d2 + (size_t)bq * KNN_K;
    const _Float16* H = h + (size_t)bq * KNN_K * ldh + c;
#pragma unroll 4
    for (int j = 0; j < KNN_K; ++j) {
        if (D[j] <= r2) m = fmaxf(m, (float)H[(size_t)j * ldh]);
    }
    xout[(size_t)bq * cout + c] = (_Float16)m;
}

// A[b*128+i][:] = [ x3 (128) | q3 pos (3) | pad ]   (row stride cinp=160)
__global__ void k_build_sa4(const _Float16* __restrict__ x3, const float* __restrict__ q3,
                            _Float16* __restrict__ A, int cinp)
{
    int gid = blockIdx.x * blockDim.x + threadIdx.x;
    if (gid >= BATCH * NS3) return;
    _Float16* R = A + (size_t)gid * cinp;
    const _Float16* X = x3 + (size_t)gid * 128;
    for (int c = 0; c < 128; ++c) R[c] = X[c];
    const float* P = q3 + (size_t)gid * 3;
#pragma unroll
    for (int d = 0; d < 3; ++d) R[128 + d] = (_Float16)P[d];
    for (int c = 131; c < cinp; ++c) R[c] = (_Float16)0.f;
}

// global max pool over the 128 points of each cloud: g[b][c]
__global__ void k_global_pool(const _Float16* __restrict__ h, _Float16* __restrict__ g)
{
    int gid = blockIdx.x * blockDim.x + threadIdx.x;
    if (gid >= BATCH * 128) return;
    int c = gid % 128, b = gid / 128;
    float m = -INFINITY;
    for (int i = 0; i < NS3; ++i)
        m = fmaxf(m, (float)h[((size_t)b * NS3 + i) * 128 + c]);
    g[gid] = (_Float16)m;
}

// A[b*128+i][:] = [ g[b] (128) | x3 (128) ]   (cin = 256, no pad)
__global__ void k_build_fp4(const _Float16* __restrict__ g, const _Float16* __restrict__ x3,
                            _Float16* __restrict__ A)
{
    int gid = blockIdx.x * blockDim.x + threadIdx.x;
    if (gid >= BATCH * NS3) return;
    int b = gid / NS3;
    _Float16* R = A + (size_t)gid * 256;
    const _Float16* G = g + (size_t)b * 128;
    const _Float16* X = x3 + (size_t)gid * 128;
    for (int c = 0; c < 128; ++c) { R[c] = G[c]; R[128 + c] = X[c]; }
}

// knn-interpolate (k=3, inverse-squared-distance) + concat skip, row stride cinp
__global__ void k_interp(const float* __restrict__ tpos, int nt,
                         const float* __restrict__ spos, int ns,
                         const _Float16* __restrict__ xsrc, int cs,
                         const _Float16* __restrict__ skip, int ck,
                         _Float16* __restrict__ A, int cinp)
{
    int gid = blockIdx.x * blockDim.x + threadIdx.x;
    if (gid >= BATCH * nt) return;
    int b = gid / nt;
    const float* T = tpos + (size_t)gid * 3;
    const float* P = spos + (size_t)b * ns * 3;
    float tx = T[0], ty = T[1], tz = T[2];
    float d0 = 1e30f, d1 = 1e30f, d2v = 1e30f;
    int i0 = 0, i1 = 0, i2 = 0;
    for (int s = 0; s < ns; ++s) {
        float dx = P[3*s] - tx, dy = P[3*s+1] - ty, dz = P[3*s+2] - tz;
        float d = dx*dx + dy*dy + dz*dz;
        if (d < d0)      { d2v = d1; i2 = i1; d1 = d0; i1 = i0; d0 = d; i0 = s; }
        else if (d < d1) { d2v = d1; i2 = i1; d1 = d; i1 = s; }
        else if (d < d2v){ d2v = d; i2 = s; }
    }
    float w0 = 1.f / fmaxf(d0, 1e-16f);
    float w1 = 1.f / fmaxf(d1, 1e-16f);
    float w2 = 1.f / fmaxf(d2v, 1e-16f);
    float wsum = w0 + w1 + w2;
    const _Float16* X0 = xsrc + ((size_t)b * ns + i0) * cs;
    const _Float16* X1 = xsrc + ((size_t)b * ns + i1) * cs;
    const _Float16* X2 = xsrc + ((size_t)b * ns + i2) * cs;
    _Float16* R = A + (size_t)gid * cinp;
    for (int c = 0; c < cs; ++c) {
        float v = (w0 * (float)X0[c] + w1 * (float)X1[c] + w2 * (float)X2[c]) / wsum;
        R[c] = (_Float16)v;
    }
    const _Float16* S = skip + (size_t)gid * ck;
    for (int c = 0; c < ck; ++c) R[cs + c] = S[c];
    for (int c = cs + ck; c < cinp; ++c) R[c] = (_Float16)0.f;
}

// transpose scores (B,N,NC)->(B,NC,N) and append softmax probabilities
__global__ void k_output(const float* __restrict__ scoresf, float* __restrict__ out)
{
    int gid = blockIdx.x * blockDim.x + threadIdx.x;
    if (gid >= BATCH * NPTS) return;
    int n = gid % NPTS, b = gid / NPTS;
    const float* s = scoresf + (size_t)gid * NCLS;
    float* so = out;
    float* po = out + (size_t)BATCH * NCLS * NPTS;
    float mx = s[0];
#pragma unroll
    for (int c = 1; c < NCLS; ++c) mx = fmaxf(mx, s[c]);
    float e[NCLS]; float sum = 0.f;
#pragma unroll
    for (int c = 0; c < NCLS; ++c) { e[c] = expf(s[c] - mx); sum += e[c]; }
#pragma unroll
    for (int c = 0; c < NCLS; ++c) {
        size_t o = ((size_t)b * NCLS + c) * NPTS + n;
        so[o] = s[c];
        po[o] = e[c] / sum;
    }
}

// ---------------- host launch ----------------

#define IN_F(i) ((const float*)d_in[(i)])

static void mlp_layer(const _Float16* A, int M, int cin, int cout,
                      const float* W, const float* b, const float* g, const float* be,
                      int mode, _Float16* outh, float* outf,
                      _Float16* w16, hipStream_t stream)
{
    int cinp = pad32(cin), coutp = pad32(cout);
    int nPk = (cinp / 32) * coutp * 32;
    k_cvt_pack<<<cdiv(nPk, 256), 256, 0, stream>>>(W, w16, cin, cout, coutp, nPk);
    int waves = (M / 16) * (coutp / 16);
    k_gemm<<<cdiv(waves, 8), 256, 0, stream>>>(A, w16, b, g, be, outh, outf,
                                               M, cinp, cout, coutp, mode);
}

extern "C" void kernel_launch(void* const* d_in, const int* in_sizes, int n_in,
                              void* d_out, int out_size, void* d_ws, size_t ws_size,
                              hipStream_t stream)
{
    (void)in_sizes; (void)n_in; (void)out_size; (void)ws_size;
    char* w = (char*)d_ws;
    size_t off = 0;
    auto alloc = [&](size_t bytes) -> void* {
        void* p = w + off;
        off = (off + bytes + 255) & ~(size_t)255;
        return p;
    };

    float*     pos   = (float*)    alloc((size_t)BATCH * NPTS * 3 * 4);
    _Float16*  feat  = (_Float16*) alloc((size_t)BATCH * NPTS * NIFC * 2);
    float*     q1    = (float*)    alloc((size_t)BATCH * NS1 * 3 * 4);
    float*     q2    = (float*)    alloc((size_t)BATCH * NS2 * 3 * 4);
    float*     q3    = (float*)    alloc((size_t)BATCH * NS3 * 3 * 4);
    int*       idx   = (int*)      alloc((size_t)BATCH * NS1 * 4);
    _Float16*  x1    = (_Float16*) alloc((size_t)BATCH * NS1 * 32 * 2);
    _Float16*  x2    = (_Float16*) alloc((size_t)BATCH * NS2 * 64 * 2);
    _Float16*  x3    = (_Float16*) alloc((size_t)BATCH * NS3 * 128 * 2);
    _Float16*  gvec  = (_Float16*) alloc((size_t)BATCH * 128 * 2);
    _Float16*  h4    = (_Float16*) alloc((size_t)BATCH * NS3 * 128 * 2);
    _Float16*  h3    = (_Float16*) alloc((size_t)BATCH * NS2 * 64 * 2);
    _Float16*  h2    = (_Float16*) alloc((size_t)BATCH * NS1 * 32 * 2);
    _Float16*  h1    = (_Float16*) alloc((size_t)BATCH * NPTS * 32 * 2);
    _Float16*  ybuf  = (_Float16*) alloc((size_t)BATCH * NPTS * 32 * 2);
    float*     scf   = (float*)    alloc((size_t)BATCH * NPTS * NCLS * 4);
    int*       nbr   = (int*)      alloc((size_t)BATCH * NS1 * KNN_K * 4);
    float*     nd2   = (float*)    alloc((size_t)BATCH * NS1 * KNN_K * 4);
    _Float16*  Abuf  = (_Float16*) alloc((size_t)BATCH * NS1 * KNN_K * 32 * 2);
    _Float16*  Hid   = (_Float16*) alloc((size_t)BATCH * NS1 * KNN_K * 32 * 2);
    _Float16*  Hb    = (_Float16*) alloc((size_t)BATCH * NS1 * KNN_K * 32 * 2);
    _Float16*  w16   = (_Float16*) alloc((size_t)32768 * 2);

    const float* cloud = IN_F(0);

    // extract positions / input features
    k_prepare<<<cdiv(BATCH * NPTS, 256), 256, 0, stream>>>(cloud, pos, feat);

    // ---------------- SA1: 8192 -> 2048, r=0.2, cin=10 ----------------
    k_fps<<<BATCH, 1024, 0, stream>>>(pos, NPTS, NS1, idx);
    k_gather<<<cdiv(BATCH * NS1, 256), 256, 0, stream>>>(pos, idx, NPTS, NS1, q1);
    k_knn<<<cdiv(BATCH * NS1, 256), 256, 0, stream>>>(q1, pos, NS1, NPTS, nbr, nd2);
    k_build_msg<<<cdiv(BATCH * NS1 * KNN_K, 256), 256, 0, stream>>>(
        feat, NIFC, pos, q1, nbr, NS1, NPTS, Abuf, 32);
    {
        int M = BATCH * NS1 * KNN_K;
        mlp_layer(Abuf, M, 10, 16, IN_F(1), IN_F(2), IN_F(3), IN_F(4), 2, Hid, nullptr, w16, stream);
        mlp_layer(Hid,  M, 16, 32, IN_F(5), IN_F(6), IN_F(7), IN_F(8), 2, Hb,  nullptr, w16, stream);
    }
    k_sa_pool<<<cdiv(BATCH * NS1 * 32, 256), 256, 0, stream>>>(Hb, nd2, 0.04f, NS1, 32, 32, x1);

    // ---------------- SA2: 2048 -> 512, r=0.4, cin=35 ----------------
    k_fps<<<BATCH, 1024, 0, stream>>>(q1, NS1, NS2, idx);
    k_gather<<<cdiv(BATCH * NS2, 256), 256, 0, stream>>>(q1, idx, NS1, NS2, q2);
    k_knn<<<cdiv(BATCH * NS2, 256), 256, 0, stream>>>(q2, q1, NS2, NS1, nbr, nd2);
    k_build_msg<<<cdiv(BATCH * NS2 * KNN_K, 256), 256, 0, stream>>>(
        x1, 32, q1, q2, nbr, NS2, NS1, Abuf, 64);
    {
        int M = BATCH * NS2 * KNN_K;
        mlp_layer(Abuf, M, 35, 32, IN_F(9),  IN_F(10), IN_F(11), IN_F(12), 2, Hid, nullptr, w16, stream);
        mlp_layer(Hid,  M, 32, 64, IN_F(13), IN_F(14), IN_F(15), IN_F(16), 2, Hb,  nullptr, w16, stream);
    }
    k_sa_pool<<<cdiv(BATCH * NS2 * 64, 256), 256, 0, stream>>>(Hb, nd2, 0.16f, NS2, 64, 64, x2);

    // ---------------- SA3: 512 -> 128, r=0.8, cin=67 ----------------
    k_fps<<<BATCH, 1024, 0, stream>>>(q2, NS2, NS3, idx);
    k_gather<<<cdiv(BATCH * NS3, 256), 256, 0, stream>>>(q2, idx, NS2, NS3, q3);
    k_knn<<<cdiv(BATCH * NS3, 256), 256, 0, stream>>>(q3, q2, NS3, NS2, nbr, nd2);
    k_build_msg<<<cdiv(BATCH * NS3 * KNN_K, 256), 256, 0, stream>>>(
        x2, 64, q2, q3, nbr, NS3, NS2, Abuf, 96);
    {
        int M = BATCH * NS3 * KNN_K;
        mlp_layer(Abuf, M, 67, 64,  IN_F(17), IN_F(18), IN_F(19), IN_F(20), 2, Hid, nullptr, w16, stream);
        mlp_layer(Hid,  M, 64, 128, IN_F(21), IN_F(22), IN_F(23), IN_F(24), 2, Hb,  nullptr, w16, stream);
    }
    k_sa_pool<<<cdiv(BATCH * NS3 * 128, 256), 256, 0, stream>>>(Hb, nd2, 0.64f, NS3, 128, 128, x3);

    // ---------------- SA4 (global): MLP([x3|p3]) -> max ----------------
    k_build_sa4<<<cdiv(BATCH * NS3, 256), 256, 0, stream>>>(x3, q3, Abuf, 160);
    {
        int M = BATCH * NS3;
        mlp_layer(Abuf, M, 131, 128, IN_F(25), IN_F(26), IN_F(27), IN_F(28), 2, Hid, nullptr, w16, stream);
        mlp_layer(Hid,  M, 128, 128, IN_F(29), IN_F(30), IN_F(31), IN_F(32), 2, Hb,  nullptr, w16, stream);
    }
    k_global_pool<<<cdiv(BATCH * 128, 256), 256, 0, stream>>>(Hb, gvec);

    // ---------------- FP4: MLP([g | x3]) ----------------
    k_build_fp4<<<cdiv(BATCH * NS3, 256), 256, 0, stream>>>(gvec, x3, Abuf);
    {
        int M = BATCH * NS3;
        mlp_layer(Abuf, M, 256, 128, IN_F(33), IN_F(34), IN_F(35), IN_F(36), 2, Hid, nullptr, w16, stream);
        mlp_layer(Hid,  M, 128, 128, IN_F(37), IN_F(38), IN_F(39), IN_F(40), 2, h4,  nullptr, w16, stream);
    }

    // ---------------- FP3: interp(h4: p3->p2) + x2 ----------------
    k_interp<<<cdiv(BATCH * NS2, 256), 256, 0, stream>>>(q2, NS2, q3, NS3, h4, 128, x2, 64, Abuf, 192);
    {
        int M = BATCH * NS2;
        mlp_layer(Abuf, M, 192, 128, IN_F(41), IN_F(42), IN_F(43), IN_F(44), 2, Hid, nullptr, w16, stream);
        mlp_layer(Hid,  M, 128, 64,  IN_F(45), IN_F(46), IN_F(47), IN_F(48), 2, h3,  nullptr, w16, stream);
    }

    // ---------------- FP2: interp(h3: p2->p1) + x1 ----------------
    k_interp<<<cdiv(BATCH * NS1, 256), 256, 0, stream>>>(q1, NS1, q2, NS2, h3, 64, x1, 32, Abuf, 96);
    {
        int M = BATCH * NS1;
        mlp_layer(Abuf, M, 96, 64, IN_F(49), IN_F(50), IN_F(51), IN_F(52), 2, Hid, nullptr, w16, stream);
        mlp_layer(Hid,  M, 64, 32, IN_F(53), IN_F(54), IN_F(55), IN_F(56), 2, h2,  nullptr, w16, stream);
    }

    // ---------------- FP1: interp(h2: p1->pos) + feat ----------------
    k_interp<<<cdiv(BATCH * NPTS, 256), 256, 0, stream>>>(pos, NPTS, q1, NS1, h2, 32, feat, NIFC, Abuf, 64);
    {
        int M = BATCH * NPTS;
        mlp_layer(Abuf, M, 39, 32, IN_F(57), IN_F(58), IN_F(59), IN_F(60), 2, Hid, nullptr, w16, stream);
        mlp_layer(Hid,  M, 32, 32, IN_F(61), IN_F(62), IN_F(63), IN_F(64), 2, Hb,  nullptr, w16, stream);
        mlp_layer(Hb,   M, 32, 32, IN_F(65), IN_F(66), IN_F(67), IN_F(68), 2, h1,  nullptr, w16, stream);

        // lin1: relu only; lin2: plain + f32 scores
        mlp_layer(h1,   M, 32, 32,   IN_F(69), IN_F(70), nullptr, nullptr, 1, ybuf, nullptr, w16, stream);
        mlp_layer(ybuf, M, 32, NCLS, IN_F(71), IN_F(72), nullptr, nullptr, 0, nullptr, scf, w16, stream);
    }

    // ---------------- output: transpose + softmax ----------------
    k_output<<<cdiv(BATCH * NPTS, 256), 256, 0, stream>>>(scf, (float*)d_out);
}
